// ANEMultiHeadAttention_51754355917371
// MI455X (gfx1250) — matrix operand order repaired and checked by reading, not compiled, not run on hardware
//
#include <hip/hip_runtime.h>
#include <hip/hip_bf16.h>

// ---------------------------------------------------------------------------
// MHA for MI455X (gfx1250): bf16 WMMA everywhere (v_wmma_f32_16x16x32_bf16),
// f32 accumulate. Compute-bound (AI ~2000 FLOP/byte); whole working set fits
// in the 192MB L2 so fragment-granularity global loads are L2 hits.
// ---------------------------------------------------------------------------

#define BB 2
#define CC 1024
#define SS 2048
#define HH 16
#define DD 64

typedef __attribute__((ext_vector_type(16))) __bf16 v16bf;
typedef __attribute__((ext_vector_type(8)))  float  v8f;

__device__ __forceinline__ unsigned short f2bf(float f) {
  unsigned u = __float_as_uint(f);
  unsigned r = u + 0x7fffu + ((u >> 16) & 1u);   // round-to-nearest-even
  return (unsigned short)(r >> 16);
}

// A-fragment (16x32 bf16, MxK, ISA 7.12.2): lane L holds row M=L%16,
// K in {8h..8h+7} U {16+8h..16+8h+7}, h=L/16; element idx = (K/16)*8 + K%8.
// => two contiguous 16-byte loads per lane from row-major (M x ldk) memory.
__device__ __forceinline__ v16bf ld_a_frag(const unsigned short* A, int ldk,
                                           int row0, int k0, int lane) {
  int r = row0 + (lane & 15);
  int h = lane >> 4;
  const unsigned short* p = A + (size_t)r * ldk + k0 + 8 * h;
  union { v16bf v; uint4 q[2]; } u;
  u.q[0] = *(const uint4*)p;
  u.q[1] = *(const uint4*)(p + 16);
  return u.v;
}

// B-fragment (32x16 bf16, KxN): lane L holds col N=L%16,
// K in {8h..8h+7} U {16+8h..16+8h+7}, h=L/16 (same order as A-frag). With B
// stored transposed (N x K row-major) this is two 16-byte loads per lane.
__device__ __forceinline__ v16bf ld_b_frag(const unsigned short* Bt, int ldk,
                                           int row0, int k0, int lane) {
  int r = row0 + (lane & 15);
  const unsigned short* p = Bt + (size_t)r * ldk + k0 + 8 * (lane >> 4);
  union { v16bf v; uint4 q[2]; } u;
  u.q[0] = *(const uint4*)p;
  u.q[1] = *(const uint4*)(p + 16);
  return u.v;
}

__device__ __forceinline__ v8f wmma_bf16(v16bf a, v16bf b, v8f c) {
  return __builtin_amdgcn_wmma_f32_16x16x32_bf16(false, a, false, b,
                                                 (short)0, c, false, false);
}

// ---------------------------------------------------------------------------
// elementwise f32 -> bf16
__global__ void cvt_f32_bf16(const float* __restrict__ in,
                             unsigned short* __restrict__ out, int n) {
  int i = blockIdx.x * blockDim.x + threadIdx.x;
  if (i < n) out[i] = f2bf(in[i]);
}

// x (B,C,S) f32 -> xT (B,S,C) bf16 (read coalesced along s)
__global__ void transpose_x(const float* __restrict__ x,
                            unsigned short* __restrict__ xT) {
  size_t t = (size_t)blockIdx.x * blockDim.x + threadIdx.x;
  if (t >= (size_t)BB * CC * SS) return;
  int s = (int)(t % SS);
  size_t bc = t / SS;
  int c = (int)(bc % CC);
  int b = (int)(bc / CC);
  xT[((size_t)b * SS + s) * CC + c] = f2bf(x[t]);
}

// ---------------------------------------------------------------------------
// GEMM: out(M,N) = A(MxK) * Bt(NxK)^T + bias, per batch (blockIdx.z).
// 8 waves/WG; wave tile 64(M) x 32(N); D = 4x2 WMMA fragments.
// mode 0: store bf16 to (b,h,S,D)  [q/k layout]
// mode 1: store bf16 to (b,h,D,S)  [v layout]
// mode 2: store f32  to (b,C,S)    [final output]
__global__ __launch_bounds__(256) void gemm_bf16(
    const unsigned short* __restrict__ A,
    const unsigned short* __restrict__ Bt,   // batch-strided N*K
    const float* __restrict__ bias,
    void* __restrict__ out,
    int M, int N, int K, int mode) {
  int bz = blockIdx.z;
  const unsigned short* Btb = Bt + (size_t)bz * N * K;
  int lane = threadIdx.x & 31;
  int wid  = threadIdx.x >> 5;
  int m0 = blockIdx.y * 128 + (wid & 1) * 64;
  int n0 = blockIdx.x * 128 + (wid >> 1) * 32;

  v8f zero = {};
  v8f c[4][2];
  for (int i = 0; i < 4; ++i)
    for (int j = 0; j < 2; ++j) c[i][j] = zero;

  for (int k0 = 0; k0 < K; k0 += 32) {
    v16bf a[4], bb[2];
#pragma unroll
    for (int i = 0; i < 4; ++i) a[i] = ld_a_frag(A, K, m0 + 16 * i, k0, lane);
#pragma unroll
    for (int j = 0; j < 2; ++j) bb[j] = ld_b_frag(Btb, K, n0 + 16 * j, k0, lane);
    if (k0 + 32 < K) {   // global_prefetch_b8 for the next K-tile
      __builtin_prefetch(A   + (size_t)(m0 + (lane & 15)) * K + k0 + 32, 0, 1);
      __builtin_prefetch(Btb + (size_t)(n0 + (lane & 15)) * K + k0 + 32, 0, 1);
    }
#pragma unroll
    for (int i = 0; i < 4; ++i)
#pragma unroll
      for (int j = 0; j < 2; ++j) c[i][j] = wmma_bf16(a[i], bb[j], c[i][j]);
  }

  // Epilogue. D-frag: VGPR r holds M = r + 8*(lane/16), N = lane%16.
  int mh = 8 * (lane >> 4);
  int nl = lane & 15;
#pragma unroll
  for (int i = 0; i < 4; ++i)
#pragma unroll
    for (int j = 0; j < 2; ++j) {
      union { v8f v; float f[8]; } u; u.v = c[i][j];
      int ng = n0 + 16 * j + nl;
#pragma unroll
      for (int r = 0; r < 8; ++r) {
        int mg = m0 + 16 * i + mh + r;
        float val = u.f[r] + (bias ? bias[mg] : 0.0f);
        if (mode == 0) {          // (b, h, S, D) bf16
          ((unsigned short*)out)[(((size_t)bz * HH + (mg >> 6)) * SS + ng) * DD
                                 + (mg & 63)] = f2bf(val);
        } else if (mode == 1) {   // (b, h, D, S) bf16
          ((unsigned short*)out)[(((size_t)bz * HH + (mg >> 6)) * DD
                                  + (mg & 63)) * SS + ng] = f2bf(val);
        } else {                  // (b, C, S) f32
          ((float*)out)[((size_t)bz * CC + mg) * SS + ng] = val;
        }
      }
    }
}

// ---------------------------------------------------------------------------
// Flash attention: one wave per (b, h, 16-query tile). Online softmax with
// half-wave __shfl_xor reductions (wave32: D-frag rows live in 16-lane
// halves). P (f32 D-frag) -> bf16 A-frag transposed through LDS, ordered by
// wave-lockstep s_wait_dscnt.
__global__ __launch_bounds__(256) void attn_flash(
    const unsigned short* __restrict__ qT,  // (b,h,S,D)
    const unsigned short* __restrict__ kT,  // (b,h,S,D)
    const unsigned short* __restrict__ vN,  // (b,h,D,S)
    unsigned short* __restrict__ oT) {      // (b,S,C)
  __shared__ unsigned short sh[8][16 * 40]; // per-wave 16x32 P tile, stride 40
  int lane = threadIdx.x & 31;
  int wid  = threadIdx.x >> 5;
  int tile = blockIdx.x * 8 + wid;          // B*H*(S/16) = 4096 tiles
  if (tile >= BB * HH * (SS / 16)) return;
  int qt = tile & (SS / 16 - 1);
  int h  = (tile >> 7) & (HH - 1);
  int b  = tile >> 11;

  const unsigned short* qTh = qT + ((size_t)b * HH + h) * SS * DD;
  const unsigned short* kTh = kT + ((size_t)b * HH + h) * SS * DD;
  const unsigned short* vh  = vN + ((size_t)b * HH + h) * DD * SS;
  int q0 = qt * 16;

  v16bf aq0 = ld_a_frag(qTh, DD, q0, 0,  lane);
  v16bf aq1 = ld_a_frag(qTh, DD, q0, 32, lane);

  v8f zero = {};
  v8f acc[4];
  for (int j = 0; j < 4; ++j) acc[j] = zero;
  float mrow[8], lrow[8];
#pragma unroll
  for (int r = 0; r < 8; ++r) { mrow[r] = -INFINITY; lrow[r] = 0.0f; }
  const float cexp = 0.125f * 1.44269504088896f;  // D^-0.5 * log2(e)

  unsigned short* shp = sh[wid];
  int nl = lane & 15;
  int mh = 8 * (lane >> 4);

  for (int kt = 0; kt < SS / 32; ++kt) {
    int kb = kt * 32;
    // scores (16 q x 32 keys) = qT(16x64) * k(64x32): K-dim is d (2 steps)
    v8f s0 = zero, s1 = zero;
    v16bf bk;
    bk = ld_b_frag(kTh, DD, kb,      0,  lane); s0 = wmma_bf16(aq0, bk, s0);
    bk = ld_b_frag(kTh, DD, kb,      32, lane); s0 = wmma_bf16(aq1, bk, s0);
    bk = ld_b_frag(kTh, DD, kb + 16, 0,  lane); s1 = wmma_bf16(aq0, bk, s1);
    bk = ld_b_frag(kTh, DD, kb + 16, 32, lane); s1 = wmma_bf16(aq1, bk, s1);

    union { v8f v; float f[8]; } u0, u1; u0.v = s0; u1.v = s1;
    float alpha[8];
#pragma unroll
    for (int r = 0; r < 8; ++r) {
      float mx = fmaxf(u0.f[r], u1.f[r]);
#pragma unroll
      for (int m = 8; m; m >>= 1) mx = fmaxf(mx, __shfl_xor(mx, m, 32));
      float mn = fmaxf(mrow[r], mx);
      alpha[r] = exp2f((mrow[r] - mn) * cexp);
      mrow[r] = mn;
      float p0 = exp2f((u0.f[r] - mn) * cexp);
      float p1 = exp2f((u1.f[r] - mn) * cexp);
      u0.f[r] = p0; u1.f[r] = p1;
      float ps = p0 + p1;
#pragma unroll
      for (int m = 8; m; m >>= 1) ps += __shfl_xor(ps, m, 32);
      lrow[r] = lrow[r] * alpha[r] + ps;
    }
#pragma unroll
    for (int j = 0; j < 4; ++j) {
      union { v8f v; float f[8]; } ua; ua.v = acc[j];
#pragma unroll
      for (int r = 0; r < 8; ++r) ua.f[r] *= alpha[r];
      acc[j] = ua.v;
    }
    // P: f32 D-frag layout -> LDS (row-major 16x32) -> bf16 A-frag
#pragma unroll
    for (int r = 0; r < 8; ++r) {
      shp[(mh + r) * 40 + nl]      = f2bf(u0.f[r]);
      shp[(mh + r) * 40 + 16 + nl] = f2bf(u1.f[r]);
    }
    asm volatile("s_wait_dscnt 0" ::: "memory");   // wave-lockstep LDS sync
    v16bf ap = ld_a_frag(shp, 40, 0, 0, lane);
    // o^T(16q x 64d) += P(16x32) * v^T(32x64); v natural layout = B^T
#pragma unroll
    for (int j = 0; j < 4; ++j) {
      v16bf bv = ld_b_frag(vh, SS, 16 * j, kb, lane);
      acc[j] = wmma_bf16(ap, bv, acc[j]);
    }
  }

  // normalize and store o^T to (b, S, C) with C index = h*64 + d
#pragma unroll
  for (int r = 0; r < 8; ++r) {
    float inv = 1.0f / lrow[r];
    int s = q0 + mh + r;
#pragma unroll
    for (int j = 0; j < 4; ++j) {
      union { v8f v; float f[8]; } ua; ua.v = acc[j];
      oT[((size_t)b * SS + s) * CC + h * DD + 16 * j + nl] = f2bf(ua.f[r] * inv);
    }
  }
}

// ---------------------------------------------------------------------------
extern "C" void kernel_launch(void* const* d_in, const int* in_sizes, int n_in,
                              void* d_out, int out_size, void* d_ws, size_t ws_size,
                              hipStream_t stream) {
  (void)in_sizes; (void)n_in; (void)out_size; (void)ws_size;
  const float* hs = (const float*)d_in[0];
  const float* Wq = (const float*)d_in[1];
  const float* bq = (const float*)d_in[2];
  const float* Wk = (const float*)d_in[3];
  const float* Wv = (const float*)d_in[4];
  const float* bv = (const float*)d_in[5];
  const float* Wo = (const float*)d_in[6];
  const float* bo = (const float*)d_in[7];

  const size_t MB = 1u << 20;
  char* ws = (char*)d_ws;
  unsigned short* xT  = (unsigned short*)(ws + 0);        // 8 MB  (B,S,C) bf16
  unsigned short* Wqb = (unsigned short*)(ws + 8  * MB);  // 2 MB
  unsigned short* Wkb = (unsigned short*)(ws + 10 * MB);
  unsigned short* Wvb = (unsigned short*)(ws + 12 * MB);
  unsigned short* Wob = (unsigned short*)(ws + 14 * MB);
  unsigned short* qTb = (unsigned short*)(ws + 16 * MB);  // 8 MB (b,h,S,D)
  unsigned short* kTb = (unsigned short*)(ws + 24 * MB);  // 8 MB (b,h,S,D)
  unsigned short* vb  = (unsigned short*)(ws + 32 * MB);  // 8 MB (b,h,D,S)
  unsigned short* oTb = (unsigned short*)(ws + 40 * MB);  // 8 MB (b,S,C)

  // precision conversion
  cvt_f32_bf16<<<(CC * CC + 255) / 256, 256, 0, stream>>>(Wq, Wqb, CC * CC);
  cvt_f32_bf16<<<(CC * CC + 255) / 256, 256, 0, stream>>>(Wk, Wkb, CC * CC);
  cvt_f32_bf16<<<(CC * CC + 255) / 256, 256, 0, stream>>>(Wv, Wvb, CC * CC);
  cvt_f32_bf16<<<(CC * CC + 255) / 256, 256, 0, stream>>>(Wo, Wob, CC * CC);
  transpose_x<<<(BB * CC * SS + 255) / 256, 256, 0, stream>>>(hs, xT);

  dim3 grid(SS / 128, CC / 128, BB);   // (16, 8, 2)
  // projections
  gemm_bf16<<<grid, 256, 0, stream>>>(Wqb, xT, bq, qTb, CC, SS, CC, 0);
  gemm_bf16<<<grid, 256, 0, stream>>>(Wkb, xT, nullptr, kTb, CC, SS, CC, 0);
  gemm_bf16<<<grid, 256, 0, stream>>>(Wvb, xT, bv, vb,  CC, SS, CC, 1);
  // attention
  attn_flash<<<(BB * HH * (SS / 16)) / 8, 256, 0, stream>>>(qTb, kTb, vb, oTb);
  // output projection (f32 out, reference layout (B,C,1,S))
  gemm_bf16<<<grid, 256, 0, stream>>>(Wob, oTb, bo, d_out, CC, SS, CC, 2);
}
